// Agent_27178553049429
// MI455X (gfx1250) — matrix operand order repaired and checked by reading, compiled, not run on hardware
//
#include <hip/hip_runtime.h>
#include <math.h>

typedef __attribute__((ext_vector_type(16))) _Float16 v16h;
typedef __attribute__((ext_vector_type(8)))  float    v8f;

#define HH 128     // hidden size
#define DD 100     // embed size
#define KX 224     // 2*D padded to multiple of 32
#define NW 4       // waves per block
#define TPB (NW * 32)

// ---------------- WMMA helpers ----------------

__device__ __forceinline__ v8f wmma16(v16h a, v16h b, v8f c) {
    // D = A(16x32 f16) * B(32x16 f16) + C(16x16 f32)
    return __builtin_amdgcn_wmma_f32_16x16x32_f16(
        false, a, false, b, (short)0, c, false, false);
}

// A-matrix fragment from LDS (rows = 16 edges/nodes, row stride = strideK halfs).
// ISA layout: lanes 0-15 -> row M=lane, K = kk*32 + {0..7, 16..23};
//             lanes 16-31 -> row M=lane-16, K = kk*32 + {8..15, 24..31}.
__device__ __forceinline__ v16h lds_loadA(const _Float16* A, int lane, int kk, int strideK) {
    const int base = (lane & 15) * strideK + kk * 32 + ((lane >> 4) & 1) * 8;
    v16h r;
    ((uint4*)&r)[0] = *(const uint4*)(A + base);
    ((uint4*)&r)[1] = *(const uint4*)(A + base + 16);
    return r;
}

// B-matrix fragment from global f16 weights (row-major [rows][strideK]).
// Lane n (0-15) holds weight row (rowbase+n), K = kk*32 + {0..7, 16..23};
// lane n+16 holds same row, K = kk*32 + {8..15, 24..31}.
__device__ __forceinline__ v16h glb_loadB(const _Float16* W, int lane, int rowbase,
                                          int kk, int strideK) {
    const _Float16* p = W + (size_t)(rowbase + (lane & 15)) * strideK
                          + kk * 32 + ((lane >> 4) & 1) * 8;
    v16h r;
    ((uint4*)&r)[0] = ((const uint4*)p)[0];
    ((uint4*)&r)[1] = ((const uint4*)(p + 16))[0];
    return r;
}

// ---------------- ordered-int float max ----------------
__device__ __forceinline__ int f2ord(float f) {
    int i = __float_as_int(f);
    return (i >= 0) ? i : (i ^ 0x7fffffff);
}
__device__ __forceinline__ float ord2f(int i) {
    return __int_as_float((i >= 0) ? i : (i ^ 0x7fffffff));
}

// Native FP32 global atomic add (global_atomic_add_f32, not a CAS loop).
__device__ __forceinline__ void atomAddF(float* p, float v) {
    unsafeAtomicAdd(p, v);
}

// ---------------- weight convert (f32 -> f16, K padded with zeros) ----------------
__global__ void k_convert(const float* __restrict__ src, _Float16* __restrict__ dst,
                          int rows, int ks, int kd) {
    int i = blockIdx.x * blockDim.x + threadIdx.x;
    if (i >= rows * kd) return;
    int r = i / kd, c = i - r * kd;
    dst[i] = (_Float16)((c < ks) ? src[r * ks + c] : 0.0f);
}

// ---------------- workspace init ----------------
__global__ void k_init(int* __restrict__ segmax, float* __restrict__ denom,
                       float* __restrict__ agg, int nN, long long nAgg) {
    long long stride = (long long)gridDim.x * blockDim.x;
    for (long long i = (long long)blockIdx.x * blockDim.x + threadIdx.x;
         i < nAgg; i += stride) {
        agg[i] = 0.0f;
        if (i < nN) { segmax[i] = (int)0x807fffffu; denom[i] = 0.0f; }
    }
}

// ---------------- pass 1: attention logits + segment max ----------------
__global__ void __launch_bounds__(TPB)
k_logit(const int* __restrict__ head_idx, const int* __restrict__ rel_idx,
        const int* __restrict__ q_idx, const int* __restrict__ tail_idx,
        const float* __restrict__ node_emb, const float* __restrict__ rel_table,
        const _Float16* __restrict__ Ws16, const _Float16* __restrict__ Wr16,
        const _Float16* __restrict__ Wqr16,
        const float* __restrict__ b_qr, const float* __restrict__ Wa,
        const float* __restrict__ b_a,
        float* __restrict__ logit, int* __restrict__ segmax, int nE) {
    __shared__ __align__(16) _Float16 sA[NW][3][16 * HH];
    const int lane = threadIdx.x & 31;
    const int w = threadIdx.x >> 5;
    const int tile = blockIdx.x * NW + w;
    const int e0 = tile * 16;
    const bool active = (e0 < nE);

    if (active) {
        _Float16* As = sA[w][0];
        _Float16* Ar = sA[w][1];
        _Float16* Aq = sA[w][2];
        // hs gather: one float4 per lane per row (32 lanes x 4 = 128 cols)
        for (int r = 0; r < 16; ++r) {
            int e = e0 + r; if (e >= nE) e = nE - 1;
            const float* src = node_emb + (size_t)head_idx[e] * HH;
            const float4 v = *(const float4*)(src + lane * 4);
            _Float16* d = As + (r << 7) + lane * 4;
            d[0] = (_Float16)v.x; d[1] = (_Float16)v.y;
            d[2] = (_Float16)v.z; d[3] = (_Float16)v.w;
        }
        // hr, qr: 100 cols + zero pad to 128; row index is wave-uniform
        for (int r = 0; r < 16; ++r) {
            int e = e0 + r; if (e >= nE) e = nE - 1;
            const float* hr = rel_table + (size_t)rel_idx[e] * DD;
            const float* qr = rel_table + (size_t)q_idx[e] * DD;
            for (int c = lane; c < HH; c += 32) {
                float vr = (c < DD) ? hr[c] : 0.0f;
                float vq = (c < DD) ? qr[c] : 0.0f;
                Ar[(r << 7) + c] = (_Float16)vr;
                Aq[(r << 7) + c] = (_Float16)vq;
            }
        }
    }
    __syncthreads();
    if (!active) return;

    const _Float16* As = sA[w][0];
    const _Float16* Ar = sA[w][1];
    const _Float16* Aq = sA[w][2];
    const int nlo = lane & 15;
    const int hi = lane >> 4;
    float accdot[8] = {0.f, 0.f, 0.f, 0.f, 0.f, 0.f, 0.f, 0.f};

    for (int jn = 0; jn < 8; ++jn) {
        v8f acc = {0.f, 0.f, 0.f, 0.f, 0.f, 0.f, 0.f, 0.f};
#pragma unroll
        for (int kk = 0; kk < 4; ++kk)
            acc = wmma16(lds_loadA(As, lane, kk, HH), glb_loadB(Ws16, lane, jn * 16, kk, HH), acc);
#pragma unroll
        for (int kk = 0; kk < 4; ++kk)
            acc = wmma16(lds_loadA(Ar, lane, kk, HH), glb_loadB(Wr16, lane, jn * 16, kk, HH), acc);
#pragma unroll
        for (int kk = 0; kk < 4; ++kk)
            acc = wmma16(lds_loadA(Aq, lane, kk, HH), glb_loadB(Wqr16, lane, jn * 16, kk, HH), acc);
        const float bias = b_qr[jn * 16 + nlo];
        const float wl = Wa[jn * 16 + nlo];
#pragma unroll
        for (int v = 0; v < 8; ++v) {
            float p = acc[v] + bias;
            accdot[v] += (p > 0.0f ? p : 0.0f) * wl;
        }
    }
#pragma unroll
    for (int m = 1; m <= 8; m <<= 1)
#pragma unroll
        for (int v = 0; v < 8; ++v)
            accdot[v] += __shfl_xor(accdot[v], m, 32);

    if (nlo == 0) {
        const float ba = b_a[0];
#pragma unroll
        for (int v = 0; v < 8; ++v) {
            int e = e0 + hi * 8 + v;
            if (e < nE) {
                float lg = accdot[v] + ba;
                logit[e] = lg;
                atomicMax(segmax + tail_idx[e], f2ord(lg));
            }
        }
    }
}

// ---------------- pass 1.5: exp + segment denom ----------------
__global__ void k_ex(const float* __restrict__ logit, const int* __restrict__ tail_idx,
                     const int* __restrict__ segmax, float* __restrict__ ex,
                     float* __restrict__ denom, int nE) {
    int e = blockIdx.x * blockDim.x + threadIdx.x;
    if (e >= nE) return;
    int t = tail_idx[e];
    float m = ord2f(segmax[t]);
    if (!isfinite(m)) m = 0.0f;
    float v = __expf(logit[e] - m);
    ex[e] = v;
    atomAddF(denom + t, v);
}

// ---------------- pass 2: GRU message + weighted scatter ----------------
__global__ void __launch_bounds__(TPB)
k_msg(const int* __restrict__ head_idx, const int* __restrict__ rel_idx,
      const int* __restrict__ ent_idx, const int* __restrict__ tail_idx,
      const float* __restrict__ node_emb, const float* __restrict__ rel_table,
      const float* __restrict__ ent_table,
      const _Float16* __restrict__ Wih16, const _Float16* __restrict__ Whh16,
      const float* __restrict__ b_ih, const float* __restrict__ b_hh,
      const float* __restrict__ ex, const float* __restrict__ denom,
      float* __restrict__ agg, int nE) {
    __shared__ __align__(16) _Float16 sx[NW][16 * KX];
    __shared__ __align__(16) _Float16 sh[NW][16 * HH];
    __shared__ float sal[NW][16];
    __shared__ int   stl[NW][16];
    const int lane = threadIdx.x & 31;
    const int w = threadIdx.x >> 5;
    const int tile = blockIdx.x * NW + w;
    const int e0 = tile * 16;
    const bool active = (e0 < nE);

    if (active) {
        _Float16* X = sx[w];
        _Float16* Hs = sh[w];
        // hs gather: one float4 per lane per row
        for (int r = 0; r < 16; ++r) {
            int e = e0 + r; if (e >= nE) e = nE - 1;
            const float* src = node_emb + (size_t)head_idx[e] * HH;
            const float4 v = *(const float4*)(src + lane * 4);
            _Float16* d = Hs + (r << 7) + lane * 4;
            d[0] = (_Float16)v.x; d[1] = (_Float16)v.y;
            d[2] = (_Float16)v.z; d[3] = (_Float16)v.w;
        }
        // x = [he | hr | 0-pad], per-row (uniform gather indices)
        for (int r = 0; r < 16; ++r) {
            int e = e0 + r; if (e >= nE) e = nE - 1;
            const float* he = ent_table + (size_t)ent_idx[e] * DD;
            const float* hr = rel_table + (size_t)rel_idx[e] * DD;
            for (int c = lane; c < KX; c += 32) {
                float v = 0.0f;
                if (c < DD)          v = he[c];
                else if (c < 2 * DD) v = hr[c - DD];
                X[r * KX + c] = (_Float16)v;
            }
        }
        if (lane < 16) {
            int e = e0 + lane; if (e >= nE) e = nE - 1;
            int t = tail_idx[e];
            stl[w][lane] = t;
            sal[w][lane] = ex[e] / (denom[t] + 1e-6f);
        }
    }
    __syncthreads();
    if (!active) return;

    const _Float16* X = sx[w];
    const _Float16* Hs = sh[w];
    const int nlo = lane & 15;
    const int hi = lane >> 4;

    for (int jn = 0; jn < 8; ++jn) {
        v8f ar  = {0.f, 0.f, 0.f, 0.f, 0.f, 0.f, 0.f, 0.f};
        v8f az  = {0.f, 0.f, 0.f, 0.f, 0.f, 0.f, 0.f, 0.f};
        v8f axn = {0.f, 0.f, 0.f, 0.f, 0.f, 0.f, 0.f, 0.f};
        v8f ahn = {0.f, 0.f, 0.f, 0.f, 0.f, 0.f, 0.f, 0.f};
#pragma unroll
        for (int kk = 0; kk < 7; ++kk) {                  // x @ W_ih^T  (K=224)
            v16h a = lds_loadA(X, lane, kk, KX);
            ar  = wmma16(a, glb_loadB(Wih16, lane, jn * 16,          kk, KX), ar);
            az  = wmma16(a, glb_loadB(Wih16, lane, HH + jn * 16,     kk, KX), az);
            axn = wmma16(a, glb_loadB(Wih16, lane, 2 * HH + jn * 16, kk, KX), axn);
        }
#pragma unroll
        for (int kk = 0; kk < 4; ++kk) {                  // hs @ W_hh^T (K=128)
            v16h a = lds_loadA(Hs, lane, kk, HH);
            ar  = wmma16(a, glb_loadB(Whh16, lane, jn * 16,          kk, HH), ar);
            az  = wmma16(a, glb_loadB(Whh16, lane, HH + jn * 16,     kk, HH), az);
            ahn = wmma16(a, glb_loadB(Whh16, lane, 2 * HH + jn * 16, kk, HH), ahn);
        }
        const int col = jn * 16 + nlo;
        const float br  = b_ih[col] + b_hh[col];
        const float bz  = b_ih[HH + col] + b_hh[HH + col];
        const float bxn = b_ih[2 * HH + col];
        const float bhn = b_hh[2 * HH + col];
#pragma unroll
        for (int v = 0; v < 8; ++v) {
            int m = v + hi * 8;
            int e = e0 + m;
            if (e >= nE) continue;
            float r = 1.0f / (1.0f + __expf(-(ar[v] + br)));
            float z = 1.0f / (1.0f + __expf(-(az[v] + bz)));
            float n = tanhf(axn[v] + bxn + r * (ahn[v] + bhn));
            float hsv = (float)Hs[(m << 7) + col];
            float val = sal[w][m] * ((1.0f - z) * n + z * hsv);
            atomAddF(agg + (size_t)stl[w][m] * HH + col, val);
        }
    }
}

// ---------------- pass 3: out = LN(relu(agg @ Wh^T)) ----------------
__global__ void __launch_bounds__(TPB)
k_out(const float* __restrict__ agg, const _Float16* __restrict__ Wh16,
      const float* __restrict__ ln_g, const float* __restrict__ ln_b,
      float* __restrict__ out, int nN) {
    __shared__ __align__(16) _Float16 sg[NW][16 * HH];
    const int lane = threadIdx.x & 31;
    const int w = threadIdx.x >> 5;
    const int tile = blockIdx.x * NW + w;
    const int n0 = tile * 16;
    const bool active = (n0 < nN);

    if (active) {
        _Float16* G = sg[w];
        for (int r = 0; r < 16; ++r) {
            int nd = n0 + r; if (nd >= nN) nd = nN - 1;
            const float4 v = *(const float4*)(agg + (size_t)nd * HH + lane * 4);
            _Float16* d = G + (r << 7) + lane * 4;
            d[0] = (_Float16)v.x; d[1] = (_Float16)v.y;
            d[2] = (_Float16)v.z; d[3] = (_Float16)v.w;
        }
    }
    __syncthreads();
    if (!active) return;

    const _Float16* G = sg[w];
    const int nlo = lane & 15;
    const int hi = lane >> 4;
    v8f o[8];
    for (int jn = 0; jn < 8; ++jn) {
        v8f acc = {0.f, 0.f, 0.f, 0.f, 0.f, 0.f, 0.f, 0.f};
#pragma unroll
        for (int kk = 0; kk < 4; ++kk)
            acc = wmma16(lds_loadA(G, lane, kk, HH), glb_loadB(Wh16, lane, jn * 16, kk, HH), acc);
#pragma unroll
        for (int v = 0; v < 8; ++v)
            o[jn][v] = acc[v] > 0.0f ? acc[v] : 0.0f;
    }
    float s[8] = {0.f, 0.f, 0.f, 0.f, 0.f, 0.f, 0.f, 0.f};
    float sq[8] = {0.f, 0.f, 0.f, 0.f, 0.f, 0.f, 0.f, 0.f};
    for (int jn = 0; jn < 8; ++jn)
#pragma unroll
        for (int v = 0; v < 8; ++v) {
            float x = o[jn][v];
            s[v] += x;
            sq[v] += x * x;
        }
#pragma unroll
    for (int m = 1; m <= 8; m <<= 1)
#pragma unroll
        for (int v = 0; v < 8; ++v) {
            s[v]  += __shfl_xor(s[v], m, 32);
            sq[v] += __shfl_xor(sq[v], m, 32);
        }
    float mu[8], rs[8];
#pragma unroll
    for (int v = 0; v < 8; ++v) {
        mu[v] = s[v] * (1.0f / HH);
        float var = sq[v] * (1.0f / HH) - mu[v] * mu[v];
        rs[v] = rsqrtf(var + 1e-5f);
    }
    for (int jn = 0; jn < 8; ++jn) {
        const int col = jn * 16 + nlo;
        const float g = ln_g[col], b = ln_b[col];
#pragma unroll
        for (int v = 0; v < 8; ++v) {
            int nd = n0 + v + hi * 8;
            if (nd < nN)
                out[(size_t)nd * HH + col] = g * (o[jn][v] - mu[v]) * rs[v] + b;
        }
    }
}

// ---------------- host launcher ----------------
extern "C" void kernel_launch(void* const* d_in, const int* in_sizes, int n_in,
                              void* d_out, int out_size, void* d_ws, size_t ws_size,
                              hipStream_t stream) {
    const int nE = in_sizes[0];
    const int nN = out_size / HH;

    const int* head_idx = (const int*)d_in[0];
    const int* rel_idx  = (const int*)d_in[1];
    const int* ent_idx  = (const int*)d_in[2];
    const int* tail_idx = (const int*)d_in[3];
    const int* q_idx    = (const int*)d_in[4];
    const float* node_emb  = (const float*)d_in[5];
    const float* ent_table = (const float*)d_in[6];
    const float* rel_table = (const float*)d_in[7];
    const float* Ws   = (const float*)d_in[8];
    const float* Wr   = (const float*)d_in[9];
    const float* Wqr  = (const float*)d_in[10];
    const float* b_qr = (const float*)d_in[11];
    const float* Wa   = (const float*)d_in[12];
    const float* b_a  = (const float*)d_in[13];
    const float* W_ih = (const float*)d_in[14];
    const float* W_hh = (const float*)d_in[15];
    const float* b_ih = (const float*)d_in[16];
    const float* b_hh = (const float*)d_in[17];
    const float* Wh   = (const float*)d_in[18];
    const float* ln_g = (const float*)d_in[19];
    const float* ln_b = (const float*)d_in[20];
    float* out = (float*)d_out;

    char* ws = (char*)d_ws;
    size_t off = 0;
    auto take = [&](size_t bytes) -> void* {
        void* p = ws + off;
        off = (off + bytes + 255) & ~(size_t)255;
        return p;
    };
    _Float16* Ws16  = (_Float16*)take((size_t)HH * HH * 2);
    _Float16* Wr16  = (_Float16*)take((size_t)HH * HH * 2);
    _Float16* Wqr16 = (_Float16*)take((size_t)HH * HH * 2);
    _Float16* Wih16 = (_Float16*)take((size_t)3 * HH * KX * 2);
    _Float16* Whh16 = (_Float16*)take((size_t)3 * HH * HH * 2);
    _Float16* Wh16  = (_Float16*)take((size_t)HH * HH * 2);
    float* logit  = (float*)take((size_t)nE * 4);
    float* exbuf  = (float*)take((size_t)nE * 4);
    int*   segmax = (int*)take((size_t)nN * 4);
    float* denom  = (float*)take((size_t)nN * 4);
    float* agg    = (float*)take((size_t)nN * HH * 4);
    (void)ws_size; (void)n_in;

    // 1) weight conversion (tiny)
    k_convert<<<(HH * HH + 255) / 256, 256, 0, stream>>>(Ws, Ws16, HH, HH, HH);
    k_convert<<<(HH * HH + 255) / 256, 256, 0, stream>>>(Wr, Wr16, HH, DD, HH);
    k_convert<<<(HH * HH + 255) / 256, 256, 0, stream>>>(Wqr, Wqr16, HH, DD, HH);
    k_convert<<<(3 * HH * KX + 255) / 256, 256, 0, stream>>>(W_ih, Wih16, 3 * HH, 2 * DD, KX);
    k_convert<<<(3 * HH * HH + 255) / 256, 256, 0, stream>>>(W_hh, Whh16, 3 * HH, HH, HH);
    k_convert<<<(HH * HH + 255) / 256, 256, 0, stream>>>(Wh, Wh16, HH, HH, HH);

    // 2) init segmax/denom/agg
    long long nAgg = (long long)nN * HH;
    k_init<<<8192, 256, 0, stream>>>(segmax, denom, agg, nN, nAgg);

    // 3) logits + segment max
    const int ntilesE = (nE + 15) / 16;
    const int blkE = (ntilesE + NW - 1) / NW;
    k_logit<<<blkE, TPB, 0, stream>>>(head_idx, rel_idx, q_idx, tail_idx,
                                      node_emb, rel_table, Ws16, Wr16, Wqr16,
                                      b_qr, Wa, b_a, logit, segmax, nE);

    // 4) exp + denom
    k_ex<<<(nE + 255) / 256, 256, 0, stream>>>(logit, tail_idx, segmax, exbuf, denom, nE);

    // 5) GRU message + weighted scatter
    k_msg<<<blkE, TPB, 0, stream>>>(head_idx, rel_idx, ent_idx, tail_idx,
                                    node_emb, rel_table, ent_table,
                                    Wih16, Whh16, b_ih, b_hh,
                                    exbuf, denom, agg, nE);

    // 6) output transform + layernorm
    const int ntilesN = (nN + 15) / 16;
    const int blkN = (ntilesN + NW - 1) / NW;
    k_out<<<blkN, TPB, 0, stream>>>(agg, Wh16, ln_g, ln_b, out, nN);
}
